// DCHL_41652592836945
// MI455X (gfx1250) — matrix operand order repaired and checked by reading, not compiled, not run on hardware
//
#include <hip/hip_runtime.h>
#include <hip/hip_bf16.h>
#include <utility>

typedef _Float16 half_t;
typedef __attribute__((ext_vector_type(16))) _Float16 v16h;
typedef __attribute__((ext_vector_type(8)))  float    v8f;

#define D_DIM 64

union V16H { v16h v; unsigned int u[8]; };

// ------------------------- elementwise helpers -------------------------
__global__ void k_zero(float* __restrict__ p, int n) {
  int i = blockIdx.x * blockDim.x + threadIdx.x;
  if (i < n) p[i] = 0.0f;
}
__global__ void k_copy(float* __restrict__ dst, const float* __restrict__ src, int n) {
  int i = blockIdx.x * blockDim.x + threadIdx.x;
  if (i < n) dst[i] = src[i];
}
__global__ void k_addinto(float* __restrict__ dst, const float* __restrict__ src, int n) {
  int i = blockIdx.x * blockDim.x + threadIdx.x;
  if (i < n) dst[i] += src[i];
}
__global__ void k_scale(float* __restrict__ p, float s, int n) {
  int i = blockIdx.x * blockDim.x + threadIdx.x;
  if (i < n) p[i] *= s;
}
__global__ void k_f32_to_f16(const float* __restrict__ x, half_t* __restrict__ y, int n) {
  int i = blockIdx.x * blockDim.x + threadIdx.x;
  if (i < n) y[i] = (half_t)x[i];
}

// ------------------------- gate: x * sigmoid(x @ W + b) -------------------------
__global__ __launch_bounds__(64)
void k_gate(const float* __restrict__ x, const float* __restrict__ w,
            const float* __restrict__ b, float* __restrict__ out) {
  __shared__ float row[D_DIM];
  int r = blockIdx.x, c = threadIdx.x;
  row[c] = x[(size_t)r * D_DIM + c];
  __syncthreads();
  float s = b[c];
  #pragma unroll 8
  for (int k = 0; k < D_DIM; ++k) s += row[k] * w[k * D_DIM + c];
  float sig = 1.0f / (1.0f + __expf(-s));
  out[(size_t)r * D_DIM + c] = row[c] * sig;
}

// ------------------------- COO SpMM scatter (out must be pre-zeroed) -------------------------
__global__ void k_spmm_scatter(const int* __restrict__ rows, const int* __restrict__ cols,
                               const float* __restrict__ vals, const float* __restrict__ x,
                               float* __restrict__ out, int nnz) {
  int i = blockIdx.x * blockDim.x + threadIdx.x;
  if (i >= nnz) return;
  int r = rows[i], c = cols[i];
  float v = vals[i];
  const float* xr = x + (size_t)c * D_DIM;
  float* orow = out + (size_t)r * D_DIM;
  for (int d = 0; d < D_DIM; ++d) atomicAdd(&orow[d], v * xr[d]);
}

// ------------------------- rowwise L2 normalize in place -------------------------
__global__ __launch_bounds__(32)
void k_l2norm(float* __restrict__ x) {
  int r = blockIdx.x, t = threadIdx.x;
  float* row = x + (size_t)r * D_DIM;
  float a = row[t], b = row[t + 32];
  float s = a * a + b * b;
  s += __shfl_xor(s, 1, 32);
  s += __shfl_xor(s, 2, 32);
  s += __shfl_xor(s, 4, 32);
  s += __shfl_xor(s, 8, 32);
  s += __shfl_xor(s, 16, 32);
  float inv = 1.0f / fmaxf(sqrtf(s), 1e-12f);
  row[t] = a * inv;
  row[t + 32] = b * inv;
}

// ------------------------- gather user rows -------------------------
__global__ __launch_bounds__(64)
void k_gather(const float* __restrict__ src, const int* __restrict__ idx,
              float* __restrict__ dst) {
  int r = blockIdx.x, c = threadIdx.x;
  dst[(size_t)r * D_DIM + c] = src[(size_t)idx[r] * D_DIM + c];
}

// ------------------------- fused GEMM(e1 @ e2^T) + exp + row-sum -------------------------
// One wave per 16-row strip; loops all 16-col tiles; neg[m] = sum_n exp(dot/T).
__global__ __launch_bounds__(32)
void k_infonce_neg(const half_t* __restrict__ e1, const half_t* __restrict__ e2,
                   float* __restrict__ neg, int ntiles) {
  const int lane = threadIdx.x;
  const int r16 = lane & 15;
  const int grp = lane >> 4;
  const int mt = blockIdx.x;

  // Load A tile (rows mt*16 .. +15, K=0..63) per CDNA5 16-bit A layout.
  V16H a0, a1;
  const half_t* arow = e1 + (size_t)(mt * 16 + r16) * D_DIM;
  #pragma unroll
  for (int v = 0; v < 8; ++v) {
    int k0 = ((v & 4) ? 16 : 0) + grp * 8 + (v & 3) * 2;
    a0.u[v] = *(const unsigned int*)(arow + k0);
    a1.u[v] = *(const unsigned int*)(arow + 32 + k0);
  }

  float acc[8];
  #pragma unroll
  for (int r = 0; r < 8; ++r) acc[r] = 0.0f;
  const float invT = 5.0f;  // 1/TEMP

  for (int nt = 0; nt < ntiles; ++nt) {
    // B tile: lane = N, VGPR v half h -> K = (v>=4?16:0) + 8*grp + 2*(v&3) + h (B = e2^T, e2 row-major)
    V16H b0, b1;
    const half_t* brow = e2 + (size_t)(nt * 16 + r16) * D_DIM;
    #pragma unroll
    for (int v = 0; v < 8; ++v) {
      int k0 = ((v & 4) ? 16 : 0) + grp * 8 + (v & 3) * 2;
      b0.u[v] = *(const unsigned int*)(brow + k0);
      b1.u[v] = *(const unsigned int*)(brow + 32 + k0);
    }
    v8f c = {};
    c = __builtin_amdgcn_wmma_f32_16x16x32_f16(false, a0.v, false, b0.v,
                                               (short)0, c, false, false);
    c = __builtin_amdgcn_wmma_f32_16x16x32_f16(false, a1.v, false, b1.v,
                                               (short)0, c, false, false);
    #pragma unroll
    for (int r = 0; r < 8; ++r) acc[r] += __expf(c[r] * invT);
  }

  // C/D layout: VGPR r holds M = r + 8*grp, N = lane&15 -> reduce over low 4 lane bits.
  #pragma unroll
  for (int r = 0; r < 8; ++r) {
    float v = acc[r];
    v += __shfl_xor(v, 1, 32);
    v += __shfl_xor(v, 2, 32);
    v += __shfl_xor(v, 4, 32);
    v += __shfl_xor(v, 8, 32);
    acc[r] = v;
  }
  if (r16 == 0) {
    int mbase = mt * 16 + grp * 8;
    #pragma unroll
    for (int r = 0; r < 8; ++r) neg[mbase + r] = acc[r];
  }
}

// ------------------------- pos + log + mean into loss scalar -------------------------
__global__ void k_infonce_loss(const float* __restrict__ e1, const float* __restrict__ e2,
                               const float* __restrict__ neg, float* __restrict__ loss, int n) {
  int m = blockIdx.x * blockDim.x + threadIdx.x;
  if (m >= n) return;
  const float* r1 = e1 + (size_t)m * D_DIM;
  const float* r2 = e2 + (size_t)m * D_DIM;
  float dot = 0.0f;
  #pragma unroll 8
  for (int c = 0; c < D_DIM; ++c) dot += r1[c] * r2[c];
  float pos = __expf(dot * 5.0f);
  float term = -__logf(pos / (neg[m] + 1e-8f) + 1e-8f);
  atomicAdd(loss, term / (float)n);
}

// ------------------------- gated fusion -------------------------
__global__ __launch_bounds__(32)
void k_fusion(const float* __restrict__ hg, const float* __restrict__ ge,
              const float* __restrict__ tr,
              const float* __restrict__ wh, const float* __restrict__ bh,
              const float* __restrict__ wg, const float* __restrict__ bg,
              const float* __restrict__ wt, const float* __restrict__ bt,
              float* __restrict__ out) {
  int r = blockIdx.x, t = threadIdx.x;
  const float* h = hg + (size_t)r * D_DIM;
  const float* g = ge + (size_t)r * D_DIM;
  const float* q = tr + (size_t)r * D_DIM;
  float h0 = h[t], h1 = h[t + 32];
  float g0 = g[t], g1 = g[t + 32];
  float q0 = q[t], q1 = q[t + 32];
  float dh = h0 * wh[t] + h1 * wh[t + 32];
  float dg = g0 * wg[t] + g1 * wg[t + 32];
  float dq = q0 * wt[t] + q1 * wt[t + 32];
  for (int m = 1; m < 32; m <<= 1) {
    dh += __shfl_xor(dh, m, 32);
    dg += __shfl_xor(dg, m, 32);
    dq += __shfl_xor(dq, m, 32);
  }
  float ch = 1.0f / (1.0f + __expf(-(dh + bh[0])));
  float cg = 1.0f / (1.0f + __expf(-(dg + bg[0])));
  float cq = 1.0f / (1.0f + __expf(-(dq + bt[0])));
  out[(size_t)r * D_DIM + t]      = ch * h0 + cg * g0 + cq * q0;
  out[(size_t)r * D_DIM + t + 32] = ch * h1 + cg * g1 + cq * q1;
}

// ------------------------- host orchestration -------------------------
extern "C" void kernel_launch(void* const* d_in, const int* in_sizes, int n_in,
                              void* d_out, int out_size, void* d_ws, size_t ws_size,
                              hipStream_t stream) {
  const int P = 12000, U = 8000, H = 12000, B = 4096;
  const int PD = P * D_DIM, UD = U * D_DIM, HD = H * D_DIM, BD = B * D_DIM;

  const float* poi_emb = (const float*)d_in[0];
  const float* w_geo   = (const float*)d_in[1];
  const float* b_geo   = (const float*)d_in[2];
  const float* w_seq   = (const float*)d_in[3];
  const float* b_seq   = (const float*)d_in[4];
  const float* w_col   = (const float*)d_in[5];
  const float* b_col   = (const float*)d_in[6];
  const float* hyper_w = (const float*)d_in[7];
  const float* hyper_b = (const float*)d_in[8];
  const float* gcn_w   = (const float*)d_in[9];
  const float* gcn_b   = (const float*)d_in[10];
  const float* trans_w = (const float*)d_in[11];
  const float* trans_b = (const float*)d_in[12];
  const float* up_vals  = (const float*)d_in[13];
  const float* pu_vals  = (const float*)d_in[14];
  const float* geo_vals = (const float*)d_in[15];
  const float* src_vals = (const float*)d_in[16];
  const float* tar_vals = (const float*)d_in[17];
  const int* up_rows  = (const int*)d_in[18];
  const int* up_cols  = (const int*)d_in[19];
  const int* pu_rows  = (const int*)d_in[20];
  const int* pu_cols  = (const int*)d_in[21];
  const int* geo_rows = (const int*)d_in[22];
  const int* geo_cols = (const int*)d_in[23];
  const int* src_rows = (const int*)d_in[24];
  const int* src_cols = (const int*)d_in[25];
  const int* tar_rows = (const int*)d_in[26];
  const int* tar_cols = (const int*)d_in[27];
  const int* user_idx = (const int*)d_in[28];

  const int nnz_up = in_sizes[13], nnz_pu = in_sizes[14], nnz_geo = in_sizes[15];
  const int nnz_src = in_sizes[16], nnz_tar = in_sizes[17];

  float* out = (float*)d_out;

  // workspace carve
  char* wsb = (char*)d_ws;
  size_t off = 0;
  auto alloc_f = [&](int n) -> float* {
    float* p = (float*)(wsb + off);
    off += (((size_t)n * 4) + 255) & ~(size_t)255;
    return p;
  };
  auto alloc_h = [&](int n) -> half_t* {
    half_t* p = (half_t*)(wsb + off);
    off += (((size_t)n * 2) + 255) & ~(size_t)255;
    return p;
  };
  float* g_col    = alloc_f(PD);
  float* g_geo    = alloc_f(PD);
  float* g_seq    = alloc_f(PD);
  float* pois_hg  = alloc_f(PD);
  float* pois_geo = alloc_f(PD);
  float* pois_tr  = alloc_f(PD);
  float* xbuf     = alloc_f(PD);
  float* ybuf     = alloc_f(PD);
  float* tmpU     = alloc_f(UD);
  float* tmpH     = alloc_f(HD);
  float* bu_hg    = alloc_f(BD);
  float* bu_geo   = alloc_f(BD);
  float* bu_tr    = alloc_f(BD);
  float* negbuf   = alloc_f(P);
  half_t* h_hgp  = alloc_h(PD);
  half_t* h_geop = alloc_h(PD);
  half_t* h_trp  = alloc_h(PD);
  half_t* h_hgu  = alloc_h(BD);
  half_t* h_geou = alloc_h(BD);
  half_t* h_tru  = alloc_h(BD);

  auto gzero = [&](float* p, int n) {
    k_zero<<<dim3((n + 255) / 256), dim3(256), 0, stream>>>(p, n);
  };
  auto gcopy = [&](float* dst, const float* src, int n) {
    k_copy<<<dim3((n + 255) / 256), dim3(256), 0, stream>>>(dst, src, n);
  };
  auto gadd = [&](float* dst, const float* src, int n) {
    k_addinto<<<dim3((n + 255) / 256), dim3(256), 0, stream>>>(dst, src, n);
  };
  auto spmm = [&](const int* r, const int* c, const float* v, int nnz,
                  const float* x, float* o, int nrows) {
    gzero(o, nrows * D_DIM);
    k_spmm_scatter<<<dim3((nnz + 255) / 256), dim3(256), 0, stream>>>(r, c, v, x, o, nnz);
  };

  // ---- gates ----
  k_gate<<<dim3(P), dim3(64), 0, stream>>>(poi_emb, w_col, b_col, g_col);
  k_gate<<<dim3(P), dim3(64), 0, stream>>>(poi_emb, w_geo, b_geo, g_geo);
  k_gate<<<dim3(P), dim3(64), 0, stream>>>(poi_emb, w_seq, b_seq, g_seq);

  // ---- conv stacks (2 residual layers, mean of [x0,x1,x2]) ----
  // hyper branch: step = PU @ (UP @ x)
  {
    float* xp = xbuf; float* yp = ybuf;
    gcopy(xp, g_col, PD);
    gcopy(pois_hg, g_col, PD);
    for (int l = 0; l < 2; ++l) {
      spmm(up_rows, up_cols, up_vals, nnz_up, xp, tmpU, U);
      spmm(pu_rows, pu_cols, pu_vals, nnz_pu, tmpU, yp, P);
      gadd(yp, xp, PD);
      std::swap(xp, yp);
      gadd(pois_hg, xp, PD);
    }
    k_scale<<<dim3((PD + 255) / 256), dim3(256), 0, stream>>>(pois_hg, 1.0f / 3.0f, PD);
  }
  // geo branch: step = GEO @ x
  {
    float* xp = xbuf; float* yp = ybuf;
    gcopy(xp, g_geo, PD);
    gcopy(pois_geo, g_geo, PD);
    for (int l = 0; l < 2; ++l) {
      spmm(geo_rows, geo_cols, geo_vals, nnz_geo, xp, yp, P);
      gadd(yp, xp, PD);
      std::swap(xp, yp);
      gadd(pois_geo, xp, PD);
    }
    k_scale<<<dim3((PD + 255) / 256), dim3(256), 0, stream>>>(pois_geo, 1.0f / 3.0f, PD);
  }
  // directed branch: step = SRC @ (TAR @ x)
  {
    float* xp = xbuf; float* yp = ybuf;
    gcopy(xp, g_seq, PD);
    gcopy(pois_tr, g_seq, PD);
    for (int l = 0; l < 2; ++l) {
      spmm(tar_rows, tar_cols, tar_vals, nnz_tar, xp, tmpH, H);
      spmm(src_rows, src_cols, src_vals, nnz_src, tmpH, yp, P);
      gadd(yp, xp, PD);
      std::swap(xp, yp);
      gadd(pois_tr, xp, PD);
    }
    k_scale<<<dim3((PD + 255) / 256), dim3(256), 0, stream>>>(pois_tr, 1.0f / 3.0f, PD);
  }

  // ---- user projections (on unnormalized pois) + gather ----
  spmm(up_rows, up_cols, up_vals, nnz_up, pois_hg, tmpU, U);
  k_gather<<<dim3(B), dim3(64), 0, stream>>>(tmpU, user_idx, bu_hg);
  spmm(up_rows, up_cols, up_vals, nnz_up, pois_geo, tmpU, U);
  k_gather<<<dim3(B), dim3(64), 0, stream>>>(tmpU, user_idx, bu_geo);
  spmm(up_rows, up_cols, up_vals, nnz_up, pois_tr, tmpU, U);
  k_gather<<<dim3(B), dim3(64), 0, stream>>>(tmpU, user_idx, bu_tr);

  // ---- L2 normalize in place ----
  k_l2norm<<<dim3(P), dim3(32), 0, stream>>>(pois_hg);
  k_l2norm<<<dim3(P), dim3(32), 0, stream>>>(pois_geo);
  k_l2norm<<<dim3(P), dim3(32), 0, stream>>>(pois_tr);
  k_l2norm<<<dim3(B), dim3(32), 0, stream>>>(bu_hg);
  k_l2norm<<<dim3(B), dim3(32), 0, stream>>>(bu_geo);
  k_l2norm<<<dim3(B), dim3(32), 0, stream>>>(bu_tr);

  // ---- f16 copies for WMMA ----
  k_f32_to_f16<<<dim3((PD + 255) / 256), dim3(256), 0, stream>>>(pois_hg, h_hgp, PD);
  k_f32_to_f16<<<dim3((PD + 255) / 256), dim3(256), 0, stream>>>(pois_geo, h_geop, PD);
  k_f32_to_f16<<<dim3((PD + 255) / 256), dim3(256), 0, stream>>>(pois_tr, h_trp, PD);
  k_f32_to_f16<<<dim3((BD + 255) / 256), dim3(256), 0, stream>>>(bu_hg, h_hgu, BD);
  k_f32_to_f16<<<dim3((BD + 255) / 256), dim3(256), 0, stream>>>(bu_geo, h_geou, BD);
  k_f32_to_f16<<<dim3((BD + 255) / 256), dim3(256), 0, stream>>>(bu_tr, h_tru, BD);

  // ---- InfoNCE losses ----
  float* loss_poi = out + BD;
  float* loss_usr = out + BD + 1;
  gzero(loss_poi, 2);

  auto infonce = [&](const float* e1f, const float* e2f,
                     const half_t* e1h, const half_t* e2h, int n, float* loss) {
    k_infonce_neg<<<dim3(n / 16), dim3(32), 0, stream>>>(e1h, e2h, negbuf, n / 16);
    k_infonce_loss<<<dim3((n + 255) / 256), dim3(256), 0, stream>>>(e1f, e2f, negbuf, loss, n);
  };
  infonce(pois_hg, pois_geo, h_hgp, h_geop, P, loss_poi);
  infonce(pois_hg, pois_tr,  h_hgp, h_trp,  P, loss_poi);
  infonce(pois_geo, pois_tr, h_geop, h_trp, P, loss_poi);
  infonce(bu_hg, bu_geo, h_hgu, h_geou, B, loss_usr);
  infonce(bu_hg, bu_tr,  h_hgu, h_tru,  B, loss_usr);
  infonce(bu_geo, bu_tr, h_geou, h_tru, B, loss_usr);

  // ---- gated fusion into d_out[0 .. B*D) ----
  k_fusion<<<dim3(B), dim3(32), 0, stream>>>(bu_hg, bu_geo, bu_tr,
                                             hyper_w, hyper_b, gcn_w, gcn_b,
                                             trans_w, trans_b, out);
  (void)n_in; (void)out_size; (void)ws_size;
}
